// GINModel_23725399343497
// MI455X (gfx1250) — compile-verified, matrix operand order checked by reading, not run on hardware
//
#include <hip/hip_runtime.h>

typedef __attribute__((ext_vector_type(16))) _Float16 v16h;
typedef __attribute__((ext_vector_type(8)))  _Float16 v8h;
typedef __attribute__((ext_vector_type(8)))  float    v8f;

#define HID 128   // hidden dim == output dim

// ---------------------------------------------------------------------------
// Convert one 128x128 fp32 weight matrix W[k][n] to fp16 transposed Wt[n][k].
// 16384 elements, launch <<<64,256>>>.
__global__ void convert_wt_kernel(const float* __restrict__ W,
                                  _Float16* __restrict__ Wt) {
    int idx = blockIdx.x * blockDim.x + threadIdx.x;   // 0..16383
    int n = idx >> 7;
    int k = idx & 127;
    Wt[idx] = (_Float16)W[k * HID + n];
}

// ---------------------------------------------------------------------------
// h[n,:] = x[n] * enc_W[:] + enc_b[:]   (one wave per node, lane = 4 channels)
__global__ void encode_kernel(const float* __restrict__ x,
                              const float* __restrict__ W,
                              const float* __restrict__ b,
                              float* __restrict__ h, int N) {
    int wave = threadIdx.x >> 5, lane = threadIdx.x & 31;
    int n = blockIdx.x * 8 + wave;
    if (n >= N) return;
    int c = lane * 4;
    float xv = x[n];
    float4 w  = *(const float4*)(W + c);
    float4 bb = *(const float4*)(b + c);
    float4 r;
    r.x = xv * w.x + bb.x;
    r.y = xv * w.y + bb.y;
    r.z = xv * w.z + bb.z;
    r.w = xv * w.w + bb.w;
    *(float4*)(h + (size_t)n * HID + c) = r;
}

// ---------------------------------------------------------------------------
// agg[i] = (1 + eps[l]) * h[i]   (float4 elementwise over N*H/4)
__global__ void scale_kernel(const float* __restrict__ h,
                             const float* __restrict__ eps, int l,
                             float* __restrict__ agg, int n4) {
    int i = blockIdx.x * blockDim.x + threadIdx.x;
    if (i >= n4) return;
    float s = 1.0f + eps[l];
    float4 v = ((const float4*)h)[i];
    v.x *= s; v.y *= s; v.z *= s; v.w *= s;
    ((float4*)agg)[i] = v;
}

// ---------------------------------------------------------------------------
// Fused GINE edge message + scatter-add:
//   agg[dst] += relu(h[src] + ea0*eW[0,:] + ea1*eW[1,:] + eb)
// One wave per edge; lane handles 4 channels.
__global__ void edge_kernel(const float* __restrict__ h,
                            const int* __restrict__ ei,
                            const float* __restrict__ ea,
                            const float* __restrict__ eW,
                            const float* __restrict__ eb,
                            float* __restrict__ agg, int E) {
    int wave = threadIdx.x >> 5, lane = threadIdx.x & 31;
    int e = blockIdx.x * 8 + wave;
    if (e >= E) return;
    int src = ei[e];
    int dst = ei[E + e];
    float a0 = ea[2 * e];
    float a1 = ea[2 * e + 1];
    int c = lane * 4;
    float4 hv = *(const float4*)(h + (size_t)src * HID + c);
    float4 w0 = *(const float4*)(eW + c);
    float4 w1 = *(const float4*)(eW + HID + c);
    float4 bb = *(const float4*)(eb + c);
    float* ap = agg + (size_t)dst * HID + c;
    atomicAdd(ap + 0, fmaxf(hv.x + a0 * w0.x + a1 * w1.x + bb.x, 0.0f));
    atomicAdd(ap + 1, fmaxf(hv.y + a0 * w0.y + a1 * w1.y + bb.y, 0.0f));
    atomicAdd(ap + 2, fmaxf(hv.z + a0 * w0.z + a1 * w1.z + bb.z, 0.0f));
    atomicAdd(ap + 3, fmaxf(hv.w + a0 * w0.w + a1 * w1.w + bb.w, 0.0f));
}

// ---------------------------------------------------------------------------
__global__ void zero_kernel(float* __restrict__ p, int n4) {
    int i = blockIdx.x * blockDim.x + threadIdx.x;
    if (i < n4) ((float4*)p)[i] = make_float4(0.f, 0.f, 0.f, 0.f);
}

// pooled[batch[n]] += h[n]   (one wave per node)
__global__ void pool_kernel(const float* __restrict__ h,
                            const int* __restrict__ batch,
                            float* __restrict__ pooled, int N) {
    int wave = threadIdx.x >> 5, lane = threadIdx.x & 31;
    int n = blockIdx.x * 8 + wave;
    if (n >= N) return;
    int g = batch[n];
    int c = lane * 4;
    float4 hv = *(const float4*)(h + (size_t)n * HID + c);
    float* pp = pooled + (size_t)g * HID + c;
    atomicAdd(pp + 0, hv.x);
    atomicAdd(pp + 1, hv.y);
    atomicAdd(pp + 2, hv.z);
    atomicAdd(pp + 3, hv.w);
}

// ---------------------------------------------------------------------------
// out[rows,128] = (relu?)(A[rows,128] @ W[128,128] + bias)
// Wt is fp16, transposed: Wt[n*128+k] = W[k][n]; staged in LDS (32 KB).
// One wave per 16-row tile; 8 column tiles (8 live accumulators).
// WMMA v_wmma_f32_16x16x32_f16, K-loop of 4 -> 32 WMMAs per wave.
// ISA 16-bit fragment layout: lane%16 = M (A) / N (B); per-lane K pattern
// {base..base+7, base+16..base+23}, base = 8*(lane>=16), per 32-K chunk.
template <bool RELU>
__global__ __launch_bounds__(256) void gemm128_wmma_kernel(
    const float* __restrict__ A, const _Float16* __restrict__ Wt,
    const float* __restrict__ bias, float* __restrict__ out, int rows) {
    __shared__ __align__(16) _Float16 sW[HID * HID];
    {
        const uint4* s = (const uint4*)Wt;
        uint4* d = (uint4*)sW;
        for (int i = threadIdx.x; i < (HID * HID) / 8; i += 256) d[i] = s[i];
    }
    __syncthreads();

    int wave = threadIdx.x >> 5, lane = threadIdx.x & 31;
    int tile = blockIdx.x * 8 + wave;
    int row0 = tile * 16;
    if (row0 >= rows) return;          // wave-uniform exit; EXEC stays all-ones

    int mloc = lane & 15;              // M within tile (A) == N within tile (B/C)
    int half = lane >> 4;              // 0 or 1
    int base = half * 8;               // K sub-pattern base
    const float* rowA = A + (size_t)(row0 + mloc) * HID;

    v8f acc[8];
#pragma unroll
    for (int nt = 0; nt < 8; ++nt) {
        float bv = bias[nt * 16 + mloc];
#pragma unroll
        for (int r = 0; r < 8; ++r) acc[nt][r] = bv;
    }

#pragma unroll
    for (int kk = 0; kk < 4; ++kk) {
        int k0 = kk * 32 + base;
        float4 a0 = *(const float4*)(rowA + k0);
        float4 a1 = *(const float4*)(rowA + k0 + 4);
        float4 a2 = *(const float4*)(rowA + k0 + 16);
        float4 a3 = *(const float4*)(rowA + k0 + 20);
        v16h a;
        a[0]  = (_Float16)a0.x; a[1]  = (_Float16)a0.y;
        a[2]  = (_Float16)a0.z; a[3]  = (_Float16)a0.w;
        a[4]  = (_Float16)a1.x; a[5]  = (_Float16)a1.y;
        a[6]  = (_Float16)a1.z; a[7]  = (_Float16)a1.w;
        a[8]  = (_Float16)a2.x; a[9]  = (_Float16)a2.y;
        a[10] = (_Float16)a2.z; a[11] = (_Float16)a2.w;
        a[12] = (_Float16)a3.x; a[13] = (_Float16)a3.y;
        a[14] = (_Float16)a3.z; a[15] = (_Float16)a3.w;
#pragma unroll
        for (int nt = 0; nt < 8; ++nt) {
            int n = nt * 16 + mloc;
            const v8h* bp = (const v8h*)(sW + n * HID + k0);
            v8h blo = bp[0];           // K = k0 .. k0+7
            v8h bhi = bp[2];           // K = k0+16 .. k0+23
            v16h b;
#pragma unroll
            for (int i = 0; i < 8; ++i) { b[i] = blo[i]; b[8 + i] = bhi[i]; }
            acc[nt] = __builtin_amdgcn_wmma_f32_16x16x32_f16(
                false, a, false, b, (short)0, acc[nt], false, false);
        }
    }

    // C/D layout: VGPR r -> row = row0 + r + 8*half, col = nt*16 + lane%16
#pragma unroll
    for (int nt = 0; nt < 8; ++nt) {
        int col = nt * 16 + mloc;
#pragma unroll
        for (int r = 0; r < 8; ++r) {
            float v = acc[nt][r];
            if (RELU) v = fmaxf(v, 0.0f);
            int M = row0 + r + half * 8;
            out[(size_t)M * HID + col] = v;
        }
    }
}

// ---------------------------------------------------------------------------
extern "C" void kernel_launch(void* const* d_in, const int* in_sizes, int n_in,
                              void* d_out, int out_size, void* d_ws, size_t ws_size,
                              hipStream_t stream) {
    const int N = in_sizes[0];          // 50000
    const int E = in_sizes[1] / 2;      // 800000
    const int G = out_size / (3 * HID); // 256

    const float* enc_W  = (const float*)d_in[12];
    const float* enc_b  = (const float*)d_in[13];
    const float* edge_W = (const float*)d_in[14];  // [L,2,H]
    const float* edge_b = (const float*)d_in[15];  // [L,H]
    const float* eps    = (const float*)d_in[16];  // [L]
    const float* W1     = (const float*)d_in[17];  // [L,H,H]
    const float* b1     = (const float*)d_in[18];
    const float* W2     = (const float*)d_in[19];
    const float* b2     = (const float*)d_in[20];
    const float* fc_W   = (const float*)d_in[21];  // [H,OUT]
    const float* fc_b   = (const float*)d_in[22];

    // Workspace layout (all offsets 16B aligned)
    float* bufA   = (float*)d_ws;                       // N*H floats
    float* bufB   = bufA + (size_t)N * HID;             // N*H floats
    float* pooled = bufB + (size_t)N * HID;             // G*H floats
    _Float16* w16 = (_Float16*)(pooled + (size_t)G * HID);
    _Float16* W1t0 = w16;
    _Float16* W1t1 = w16 + 16384;
    _Float16* W2t0 = w16 + 2 * 16384;
    _Float16* W2t1 = w16 + 3 * 16384;
    _Float16* fcWt = w16 + 4 * 16384;

    // Pre-convert weights to fp16 transposed (once per launch, deterministic)
    convert_wt_kernel<<<64, 256, 0, stream>>>(W1,             W1t0);
    convert_wt_kernel<<<64, 256, 0, stream>>>(W1 + HID * HID, W1t1);
    convert_wt_kernel<<<64, 256, 0, stream>>>(W2,             W2t0);
    convert_wt_kernel<<<64, 256, 0, stream>>>(W2 + HID * HID, W2t1);
    convert_wt_kernel<<<64, 256, 0, stream>>>(fc_W,           fcWt);

    const int n4        = (N * HID) / 4;
    const int nodeBlk   = (N + 7) / 8;
    const int edgeBlk   = (E + 7) / 8;
    const int scaleBlk  = (n4 + 255) / 256;
    const int gemmBlk   = ((N + 15) / 16 + 7) / 8;
    const int gP4       = (G * HID) / 4;
    const int fcBlk     = ((G + 15) / 16 + 7) / 8;

    for (int g = 0; g < 3; ++g) {
        const float* x  = (const float*)d_in[4 * g + 0];
        const int*   ei = (const int*)  d_in[4 * g + 1];
        const float* ea = (const float*)d_in[4 * g + 2];
        const int*   bt = (const int*)  d_in[4 * g + 3];

        encode_kernel<<<nodeBlk, 256, 0, stream>>>(x, enc_W, enc_b, bufA, N);

        float* hcur = bufA;
        float* aggb = bufB;
        for (int l = 0; l < 2; ++l) {
            scale_kernel<<<scaleBlk, 256, 0, stream>>>(hcur, eps, l, aggb, n4);
            edge_kernel<<<edgeBlk, 256, 0, stream>>>(
                hcur, ei, ea, edge_W + (size_t)l * 2 * HID, edge_b + (size_t)l * HID,
                aggb, E);
            _Float16* w1t = (l == 0) ? W1t0 : W1t1;
            _Float16* w2t = (l == 0) ? W2t0 : W2t1;
            gemm128_wmma_kernel<true><<<gemmBlk, 256, 0, stream>>>(
                aggb, w1t, b1 + (size_t)l * HID, hcur, N);
            gemm128_wmma_kernel<true><<<gemmBlk, 256, 0, stream>>>(
                hcur, w2t, b2 + (size_t)l * HID, aggb, N);
            float* t = hcur; hcur = aggb; aggb = t;   // h now lives in (old) aggb
        }

        zero_kernel<<<(gP4 + 255) / 256, 256, 0, stream>>>(pooled, gP4);
        pool_kernel<<<nodeBlk, 256, 0, stream>>>(hcur, bt, pooled, N);
        gemm128_wmma_kernel<false><<<fcBlk, 256, 0, stream>>>(
            pooled, fcWt, fc_b, (float*)d_out + (size_t)g * G * HID, G);
    }
}
